// GAT_77077483094063
// MI455X (gfx1250) — matrix operand order repaired and checked by reading, not compiled, not run on hardware
//
#include <hip/hip_runtime.h>
#include <hip/hip_bf16.h>
#include <math.h>

// ---------------------------------------------------------------------------
// Types for CDNA5 WMMA (wave32): V_WMMA_F32_16X16X32_F16
// ---------------------------------------------------------------------------
typedef __attribute__((ext_vector_type(16))) _Float16 v16h;
typedef __attribute__((ext_vector_type(8)))  _Float16 v8h;
typedef __attribute__((ext_vector_type(8)))  float    v8f;

#define NNODES 50000L
#define TM 64
#define TK 32

// ---------------------------------------------------------------------------
// Tiled GEMM: C[M,N] = A[M,K] @ W[K,N] (+ bias[N])
// A: f16 row-major [M,K]; Wt: f16 row-major W^T [N,K]; f32 accumulate.
// Block: 128 threads = 4 waves; block tile 64 x (NG*16); wave w owns rows
// [w*16, w*16+16).
//
// Software-pipelined, double-buffered LDS staging using the CDNA5 async
// engine (GLOBAL_LOAD_ASYNC_TO_LDS_B128 / ASYNCcnt): tile t+1 is issued
// before waiting on tile t; async loads complete in order, so
// s_wait_asynccnt <= ASYNC_N guarantees tile t is resident while t+1 is
// still in flight under the WMMAs. Every thread issues exactly ASYNC_N
// asyncs per tile so the wait threshold is a compile-time constant.
// OOB rows/cols are address-clamped; their garbage outputs are store-guarded.
// K must be a multiple of 32 (true for all call sites: 256, 64, 512).
// ---------------------------------------------------------------------------
template <int NG>
__launch_bounds__(128)
__global__ void gemm_wmma_f16(const _Float16* __restrict__ A,
                              const _Float16* __restrict__ Wt,
                              const float* __restrict__ bias,
                              float* __restrict__ C,
                              int M, int N, int K) {
    constexpr int TN = NG * 16;
    constexpr unsigned ABYTES = TM * TK * 2;   // 4096
    constexpr unsigned BBYTES = TN * TK * 2;   // 4096 (NG=4) / 1024 (NG=1)
    __shared__ __align__(16) _Float16 As[2][TM * TK];   // [row][k] 64x32 f16
    __shared__ __align__(16) _Float16 Bs[2][TN * TK];   // [col][k] W^T tile

    const int tid  = threadIdx.x;
    const int wave = tid >> 5;
    const int lane = tid & 31;
    const int m0   = blockIdx.x * TM;
    const int n0   = blockIdx.y * TN;

    v8f acc[NG] = {};

    // ---- async staging assignments (16-byte chunks = 8 halfs) ----
    // A tile: 64 rows x 4 chunks = 256 chunks; thread handles {tid, tid+128}.
    const int ar0 = tid >> 2;                 // 0..31
    const int ar1 = ar0 + 32;                 // 32..63
    const int aseg = (tid & 3) << 3;          // 0,8,16,24 (halfs)
    const long ga0 = (long)min(m0 + ar0, M - 1) * K + aseg;
    const long ga1 = (long)min(m0 + ar1, M - 1) * K + aseg;
    const unsigned lA0 = (unsigned)(uintptr_t)&As[0][ar0 * TK + aseg];
    const unsigned lA1 = (unsigned)(uintptr_t)&As[0][ar1 * TK + aseg];
    // B tile: TN cols x 4 chunks. NG=4 -> 256 chunks (2/thread).
    // NG=1 -> 64 chunks; threads 64..127 duplicate (same bytes, same LDS
    // address) so every thread issues the same number of asyncs.
    const int bch0 = (NG == 4) ? tid : (tid & 63);
    const int bc0 = bch0 >> 2, bs0 = (bch0 & 3) << 3;
    const int bc1 = bc0 + 32;                 // NG=4 only
    const long gb0 = (long)min(n0 + bc0, N - 1) * K + bs0;
    const long gb1 = (long)min(n0 + (NG == 4 ? bc1 : 0), N - 1) * K + bs0;
    const unsigned lB0 = (unsigned)(uintptr_t)&Bs[0][(bc0 & (TN - 1)) * TK + bs0];
    const unsigned lB1 = (unsigned)(uintptr_t)&Bs[0][((NG == 4 ? bc1 : 0) & (TN - 1)) * TK + bs0];

#define ISSUE_TILE(BSEL, KOFF)                                                  \
    do {                                                                        \
        const unsigned aofs_ = (unsigned)(BSEL) * ABYTES;                       \
        const unsigned bofs_ = (unsigned)(BSEL) * BBYTES;                       \
        asm volatile("global_load_async_to_lds_b128 %0, %1, off"                \
                     :: "v"(lA0 + aofs_), "v"(A + ga0 + (KOFF)) : "memory");    \
        asm volatile("global_load_async_to_lds_b128 %0, %1, off"                \
                     :: "v"(lA1 + aofs_), "v"(A + ga1 + (KOFF)) : "memory");    \
        asm volatile("global_load_async_to_lds_b128 %0, %1, off"                \
                     :: "v"(lB0 + bofs_), "v"(Wt + gb0 + (KOFF)) : "memory");   \
        if (NG == 4)                                                            \
            asm volatile("global_load_async_to_lds_b128 %0, %1, off"            \
                         :: "v"(lB1 + bofs_), "v"(Wt + gb1 + (KOFF)) : "memory"); \
    } while (0)

    const int ntiles = K / TK;
    ISSUE_TILE(0, 0);                         // prologue: tile 0 in flight

    for (int t = 0; t < ntiles; ++t) {
        const int bsel = t & 1;
        if (t + 1 < ntiles) {
            ISSUE_TILE((t + 1) & 1, (t + 1) * TK);
            // tile t resident once count drops to the in-flight tile's quota
            // (async loads complete in order)
            if (NG == 4) asm volatile("s_wait_asynccnt 0x4" ::: "memory");
            else         asm volatile("s_wait_asynccnt 0x3" ::: "memory");
        } else {
            asm volatile("s_wait_asynccnt 0x0" ::: "memory");
        }
        __syncthreads();

        // ---- A fragment (16-bit A 16x32; this code's contiguous k order, same as B):
        //   lanes 0-15:  M=lane,    K = {0..15}
        //   lanes 16-31: M=lane-16, K = {16..31}
        const int mr = (wave << 4) + (lane & 15);
        const int kb = (lane < 16) ? 0 : 16;
        v16h a;
        *((v8h*)&a)     = *(const v8h*)&As[bsel][mr * TK + kb];
        *((v8h*)&a + 1) = *(const v8h*)&As[bsel][mr * TK + kb + 8];

        // ---- preload all B fragments, then stream the WMMAs back-to-back
        const int kc = (lane < 16) ? 0 : 16;
        v16h bfrag[NG];
        #pragma unroll
        for (int g = 0; g < NG; ++g) {
            const int col = (g << 4) + (lane & 15);
            bfrag[g] = *(const v16h*)&Bs[bsel][col * TK + kc];
        }
        #pragma unroll
        for (int g = 0; g < NG; ++g) {
            acc[g] = __builtin_amdgcn_wmma_f32_16x16x32_f16(
                /*neg_a=*/false, a, /*neg_b=*/false, bfrag[g],
                /*c_mod=*/(short)0, acc[g],
                /*reuse_a=*/false, /*reuse_b=*/false);
        }
        __syncthreads();   // all waves done reading buf[bsel] before reuse
    }
#undef ISSUE_TILE

    // ---- epilogue: C/D layout — vgpr r holds row r (lanes 0-15) / 8+r ----
    const int rbase = m0 + (wave << 4) + ((lane >> 4) << 3);
    #pragma unroll
    for (int g = 0; g < NG; ++g) {
        const int ncol = n0 + (g << 4) + (lane & 15);
        if (ncol < N) {
            const float bv = bias ? bias[ncol] : 0.0f;
            #pragma unroll
            for (int r = 0; r < 8; ++r) {
                const int row = rbase + r;
                if (row < M) C[(long)row * N + ncol] = acc[g][r] + bv;
            }
        }
    }
}

// ---------------------------------------------------------------------------
// Operand preparation: f32 -> f16 copies / transposed weights.
// ---------------------------------------------------------------------------
__global__ void f32_to_f16_kernel(const float* __restrict__ src,
                                  _Float16* __restrict__ dst, long n) {
    const long i = (long)blockIdx.x * blockDim.x + threadIdx.x;
    if (i < n) dst[i] = (_Float16)src[i];
}

// Wt[n*K + k] = (f16) W[k*N + n]; N is a power of two (64/512/16).
__global__ void transpose_f16_kernel(const float* __restrict__ W,
                                     _Float16* __restrict__ Wt,
                                     int K, int logN) {
    const int idx = blockIdx.x * blockDim.x + threadIdx.x;
    if (idx >= (K << logN)) return;
    const int n = idx & ((1 << logN) - 1);
    const int k = idx >> logN;
    Wt[(long)n * K + k] = (_Float16)W[idx];
}

// ---------------------------------------------------------------------------
// el[n,h] = sum_d ft[n,h,d]*al[h,d] ; er likewise. One thread per (n,h).
// ---------------------------------------------------------------------------
__global__ void scores_kernel(const float* __restrict__ ft,
                              const float* __restrict__ al,
                              const float* __restrict__ ar,
                              float* __restrict__ el, float* __restrict__ er,
                              int total, int hmask, int D) {
    const int idx = blockIdx.x * blockDim.x + threadIdx.x;
    if (idx >= total) return;
    const int h = idx & hmask;
    const float* f  = ft + (long)idx * D;
    const float* wl = al + h * D;
    const float* wr = ar + h * D;
    float sl = 0.0f, sr = 0.0f;
    #pragma unroll 8
    for (int d = 0; d < D; ++d) { const float v = f[d]; sl += v * wl[d]; sr += v * wr[d]; }
    el[idx] = sl;
    er[idx] = sr;
}

__global__ void init_ms_kernel(float* __restrict__ m, float* __restrict__ s, int n) {
    const int i = blockIdx.x * blockDim.x + threadIdx.x;
    if (i < n) { m[i] = -INFINITY; s[i] = 0.0f; }
}

__global__ void zero_kernel(float* __restrict__ p, long n) {
    const long i = (long)blockIdx.x * blockDim.x + threadIdx.x;
    if (i < n) p[i] = 0.0f;
}

__device__ __forceinline__ float leaky02(float x) { return x > 0.0f ? x : 0.2f * x; }

// Sign-split float atomic max (correct for mixed signs, init to -inf).
__device__ __forceinline__ void atomicMaxF(float* a, float v) {
    if (v >= 0.0f) atomicMax((int*)a, __float_as_int(v));
    else           atomicMin((unsigned int*)a, __float_as_uint(v));
}

__global__ void edge_max_kernel(const int* __restrict__ src, const int* __restrict__ dst,
                                const float* __restrict__ el, const float* __restrict__ er,
                                float* __restrict__ m, long total, int logH) {
    const long idx = (long)blockIdx.x * blockDim.x + threadIdx.x;
    if (idx >= total) return;
    const int H = 1 << logH;
    const int h = (int)(idx & (H - 1));
    const long e = idx >> logH;
    const long is = (long)src[e] * H + h;
    const long id = (long)dst[e] * H + h;
    atomicMaxF(&m[id], leaky02(el[is] + er[id]));
}

__global__ void edge_sum_kernel(const int* __restrict__ src, const int* __restrict__ dst,
                                const float* __restrict__ el, const float* __restrict__ er,
                                const float* __restrict__ m, float* __restrict__ s,
                                long total, int logH) {
    const long idx = (long)blockIdx.x * blockDim.x + threadIdx.x;
    if (idx >= total) return;
    const int H = 1 << logH;
    const int h = (int)(idx & (H - 1));
    const long e = idx >> logH;
    const long is = (long)src[e] * H + h;
    const long id = (long)dst[e] * H + h;
    atomicAdd(&s[id], expf(leaky02(el[is] + er[id]) - m[id]));
}

// One lane per (edge, head, d): coalesced ft gather + coalesced atomic scatter.
__global__ void edge_agg_kernel(const int* __restrict__ src, const int* __restrict__ dst,
                                const float* __restrict__ el, const float* __restrict__ er,
                                const float* __restrict__ m, const float* __restrict__ s,
                                const float* __restrict__ ft, float* __restrict__ rst,
                                long total, int logH, int logD) {
    const long idx = (long)blockIdx.x * blockDim.x + threadIdx.x;
    if (idx >= total) return;
    const int D = 1 << logD, H = 1 << logH;
    const int d = (int)(idx & (D - 1));
    const long eh = idx >> logD;
    const int h = (int)(eh & (H - 1));
    const long e = eh >> logH;
    const long is = (long)src[e] * H + h;
    const long id = (long)dst[e] * H + h;
    const float alpha = expf(leaky02(el[is] + er[id]) - m[id]) / s[id];
    atomicAdd(&rst[id * D + d], alpha * ft[is * D + d]);
}

// out = [elu]( rst [+ resid] + bias[j & hdmask] ); optional extra f32 and f16
// destinations (the f16 mirror feeds the next layer's WMMA GEMM).
__global__ void finalize_kernel(const float* __restrict__ rst,
                                const float* __restrict__ resid,
                                const float* __restrict__ bias,
                                float* __restrict__ dst1, float* __restrict__ dst2,
                                _Float16* __restrict__ dst16,
                                long total, int hdmask, int do_elu) {
    const long i = (long)blockIdx.x * blockDim.x + threadIdx.x;
    if (i >= total) return;
    float v = rst[i] + bias[(int)(i & hdmask)];
    if (resid) v += resid[i];
    if (do_elu) v = v > 0.0f ? v : expm1f(v);
    dst1[i] = v;
    if (dst2) dst2[i] = v;
    if (dst16) dst16[i] = (_Float16)v;
}

// ---------------------------------------------------------------------------
static inline int cdiv(long a, int b) { return (int)((a + b - 1) / b); }

extern "C" void kernel_launch(void* const* d_in, const int* in_sizes, int n_in,
                              void* d_out, int out_size, void* d_ws, size_t ws_size,
                              hipStream_t stream) {
    const float* feat0   = (const float*)d_in[0];
    const float* feat1   = (const float*)d_in[1];
    const float* fc0_w   = (const float*)d_in[2];
    const float* fc0_b   = (const float*)d_in[3];
    const float* fc1_w   = (const float*)d_in[4];
    const float* fc1_b   = (const float*)d_in[5];
    const float* l0_W    = (const float*)d_in[6];
    const float* l0_al   = (const float*)d_in[7];
    const float* l0_ar   = (const float*)d_in[8];
    const float* l0_b    = (const float*)d_in[9];
    const float* l1_W    = (const float*)d_in[10];
    const float* l1_al   = (const float*)d_in[11];
    const float* l1_ar   = (const float*)d_in[12];
    const float* l1_b    = (const float*)d_in[13];
    const float* l2_W    = (const float*)d_in[14];
    const float* l2_al   = (const float*)d_in[15];
    const float* l2_ar   = (const float*)d_in[16];
    const float* l2_b    = (const float*)d_in[17];
    const float* l2_res_w= (const float*)d_in[18];
    const int*   src     = (const int*)d_in[19];
    const int*   dst     = (const int*)d_in[20];
    const long   E       = in_sizes[19];

    float* out = (float*)d_out;                 // [0,800000): logits, then encoded
    float* enc_out = out + NNODES * 16;

    // workspace layout
    float* bufA = (float*)d_ws;                 // 50000*512 f32
    float* bufB = bufA + NNODES * 512;          // 50000*512 f32 (ft)
    float* bufC = bufB + NNODES * 512;          // 50000*512 f32 (rst / h)
    float* el   = bufC + NNODES * 512;          // 50000*8
    float* er   = el   + NNODES * 8;
    float* mB   = er   + NNODES * 8;
    float* sB   = mB   + NNODES * 8;
    float* res2 = sB   + NNODES * 8;            // 50000*16
    _Float16* a16   = (_Float16*)(res2 + NNODES * 16);  // 50000*512 halfs (activations)
    _Float16* w_fc0 = a16 + NNODES * 512;       // 64*256
    _Float16* w_fc1 = w_fc0 + 64 * 256;         // 64*256
    _Float16* w_l0  = w_fc1 + 64 * 256;         // 512*64
    _Float16* w_l1  = w_l0 + 512 * 64;          // 512*512
    _Float16* w_l2  = w_l1 + 512 * 512;         // 16*512
    _Float16* w_l2r = w_l2 + 16 * 512;          // 16*512

    const dim3 blk128(128), blk256(256);

    // ======== operand prep: transposed f16 weights, f16 feats ========
    transpose_f16_kernel<<<cdiv(256 * 64, 256), blk256, 0, stream>>>(fc0_w, w_fc0, 256, 6);
    transpose_f16_kernel<<<cdiv(256 * 64, 256), blk256, 0, stream>>>(fc1_w, w_fc1, 256, 6);
    transpose_f16_kernel<<<cdiv(64 * 512, 256), blk256, 0, stream>>>(l0_W, w_l0, 64, 9);
    transpose_f16_kernel<<<cdiv(512 * 512, 256), blk256, 0, stream>>>(l1_W, w_l1, 512, 9);
    transpose_f16_kernel<<<cdiv(512 * 16, 256), blk256, 0, stream>>>(l2_W, w_l2, 512, 4);
    transpose_f16_kernel<<<cdiv(512 * 16, 256), blk256, 0, stream>>>(l2_res_w, w_l2r, 512, 4);
    f32_to_f16_kernel<<<cdiv(25000L * 256, 256), blk256, 0, stream>>>(feat0, a16, 25000L * 256);
    f32_to_f16_kernel<<<cdiv(25000L * 256, 256), blk256, 0, stream>>>(feat1, a16 + 25000L * 256, 25000L * 256);

    // ======== input projections: h = concat(feat0@fc0+b, feat1@fc1+b) [N,64]
    gemm_wmma_f16<4><<<dim3(cdiv(25000, TM), 1), blk128, 0, stream>>>(
        a16, w_fc0, fc0_b, bufA, 25000, 64, 256);
    gemm_wmma_f16<4><<<dim3(cdiv(25000, TM), 1), blk128, 0, stream>>>(
        a16 + 25000L * 256, w_fc1, fc1_b, bufA + 25000L * 64, 25000, 64, 256);
    f32_to_f16_kernel<<<cdiv(NNODES * 64, 256), blk256, 0, stream>>>(bufA, a16, NNODES * 64);

    // ======== layer 0: 64 -> 8 heads x 64, no residual, ELU ========
    gemm_wmma_f16<4><<<dim3(cdiv(NNODES, TM), 8), blk128, 0, stream>>>(
        a16, w_l0, nullptr, bufB, (int)NNODES, 512, 64);
    scores_kernel<<<cdiv(NNODES * 8, 256), blk256, 0, stream>>>(
        bufB, l0_al, l0_ar, el, er, (int)(NNODES * 8), 7, 64);
    init_ms_kernel<<<cdiv(NNODES * 8, 256), blk256, 0, stream>>>(mB, sB, (int)(NNODES * 8));
    zero_kernel<<<cdiv(NNODES * 512, 256), blk256, 0, stream>>>(bufC, NNODES * 512);
    edge_max_kernel<<<cdiv(E * 8, 256), blk256, 0, stream>>>(src, dst, el, er, mB, E * 8, 3);
    edge_sum_kernel<<<cdiv(E * 8, 256), blk256, 0, stream>>>(src, dst, el, er, mB, sB, E * 8, 3);
    edge_agg_kernel<<<cdiv(E * 8 * 64, 256), blk256, 0, stream>>>(
        src, dst, el, er, mB, sB, bufB, bufC, E * 8 * 64, 3, 6);
    finalize_kernel<<<cdiv(NNODES * 512, 256), blk256, 0, stream>>>(
        bufC, nullptr, l0_b, bufC, nullptr, a16, NNODES * 512, 511, 1);

    // ======== layer 1: 512 -> 8 heads x 64, identity residual, ELU ========
    gemm_wmma_f16<4><<<dim3(cdiv(NNODES, TM), 8), blk128, 0, stream>>>(
        a16, w_l1, nullptr, bufB, (int)NNODES, 512, 512);
    scores_kernel<<<cdiv(NNODES * 8, 256), blk256, 0, stream>>>(
        bufB, l1_al, l1_ar, el, er, (int)(NNODES * 8), 7, 64);
    init_ms_kernel<<<cdiv(NNODES * 8, 256), blk256, 0, stream>>>(mB, sB, (int)(NNODES * 8));
    zero_kernel<<<cdiv(NNODES * 512, 256), blk256, 0, stream>>>(bufA, NNODES * 512);
    edge_max_kernel<<<cdiv(E * 8, 256), blk256, 0, stream>>>(src, dst, el, er, mB, E * 8, 3);
    edge_sum_kernel<<<cdiv(E * 8, 256), blk256, 0, stream>>>(src, dst, el, er, mB, sB, E * 8, 3);
    edge_agg_kernel<<<cdiv(E * 8 * 64, 256), blk256, 0, stream>>>(
        src, dst, el, er, mB, sB, bufB, bufA, E * 8 * 64, 3, 6);
    // encoded = elu(rst + h + b); writes: H (f32), d_out encoded, H f16 for layer 2
    finalize_kernel<<<cdiv(NNODES * 512, 256), blk256, 0, stream>>>(
        bufA, bufC, l1_b, bufA, enc_out, a16, NNODES * 512, 511, 1);

    // ======== layer 2: 512 -> 1 head x 16, linear residual, no act ========
    gemm_wmma_f16<1><<<dim3(cdiv(NNODES, TM), 1), blk128, 0, stream>>>(
        a16, w_l2, nullptr, bufB, (int)NNODES, 16, 512);
    gemm_wmma_f16<1><<<dim3(cdiv(NNODES, TM), 1), blk128, 0, stream>>>(
        a16, w_l2r, nullptr, res2, (int)NNODES, 16, 512);
    scores_kernel<<<cdiv(NNODES, 256), blk256, 0, stream>>>(
        bufB, l2_al, l2_ar, el, er, (int)NNODES, 0, 16);
    init_ms_kernel<<<cdiv(NNODES, 256), blk256, 0, stream>>>(mB, sB, (int)NNODES);
    zero_kernel<<<cdiv(NNODES * 16, 256), blk256, 0, stream>>>(bufC, NNODES * 16);
    edge_max_kernel<<<cdiv(E, 256), blk256, 0, stream>>>(src, dst, el, er, mB, E, 0);
    edge_sum_kernel<<<cdiv(E, 256), blk256, 0, stream>>>(src, dst, el, er, mB, sB, E, 0);
    edge_agg_kernel<<<cdiv(E * 16, 256), blk256, 0, stream>>>(
        src, dst, el, er, mB, sB, bufB, bufC, E * 16, 0, 4);
    finalize_kernel<<<cdiv(NNODES * 16, 256), blk256, 0, stream>>>(
        bufC, res2, l2_b, out, nullptr, nullptr, NNODES * 16, 15, 0);
}
